// GNN_85005992723091
// MI455X (gfx1250) — matrix operand order repaired and checked by reading, not compiled, not run on hardware
//
#include <hip/hip_runtime.h>

#define N_NODES 20000
#define N_EDGES 640000
#define D 128
#define D2 256
#define NODE_IN 74
#define EDGE_IN 12
#define NLAYERS 5
#define BN_EPS 1e-5f

typedef _Float16 half_t;
typedef _Float16 v16h __attribute__((ext_vector_type(16)));
typedef float    v8f  __attribute__((ext_vector_type(8)));

// Dynamic LDS layout for the WMMA MLP kernel:
//   w1t : f16 [256][128]  (W1 transposed)   65536 B
//   w2t : f16 [128][256]  (W2 transposed)   65536 B
//   b1s : f32 [256]                          1024 B
//   b2s : f32 [128]                           512 B
//   hid : f16  8 waves x [16][256]          65536 B
#define MLP_SMEM_BYTES (65536 + 65536 + 1024 + 512 + 65536)

// ---------------------------------------------------------------------------
// h0 = relu(x @ Wn + b); also initializes the JK accumulator (d_out) with h0.
// One block per node, 128 threads = 128 output channels.
// ---------------------------------------------------------------------------
__global__ void __launch_bounds__(128) node_proj_kernel(
    const float* __restrict__ x, const float* __restrict__ w,
    const float* __restrict__ b, float* __restrict__ h,
    float* __restrict__ jk) {
  __shared__ float xr[NODE_IN];
  const int node = blockIdx.x;
  const int c = threadIdx.x;
  if (c < NODE_IN) xr[c] = x[node * NODE_IN + c];
  __syncthreads();
  float acc = b[c];
#pragma unroll 2
  for (int k = 0; k < NODE_IN; ++k) acc += xr[k] * w[k * D + c];
  acc = fmaxf(acc, 0.f);
  h[(size_t)node * D + c] = acc;
  jk[(size_t)node * D + c] = acc;
}

__global__ void zero_kernel(float* __restrict__ p, int n) {
  int i = blockIdx.x * 256 + threadIdx.x;
  if (i < n) p[i] = 0.f;
}

// ---------------------------------------------------------------------------
// Edge phase, fused: e = relu(edge_attr @ We + be) recomputed on the fly
// (avoids a 327 MB e-buffer), msg = h[src] + e, atomicAdd into agg[dst].
// Block = 256 threads = 8 edges x 32 channel-groups (4 channels each).
// agg/h are L2-resident (10 MB each) so atomics/gathers stay on-chip.
// ---------------------------------------------------------------------------
__global__ void __launch_bounds__(256) edge_scatter_kernel(
    const int* __restrict__ ei, const float* __restrict__ eattr,
    const float* __restrict__ we, const float* __restrict__ eb,
    const float* __restrict__ h, float* __restrict__ agg) {
  __shared__ float wS[EDGE_IN * D];
  __shared__ float bS[D];
  __shared__ float aS[8 * EDGE_IN];
  const int t = threadIdx.x;
  for (int i = t; i < EDGE_IN * D; i += 256) wS[i] = we[i];
  if (t < D) bS[t] = eb[t];
  const int eBase = blockIdx.x * 8;
  if (t < 8 * EDGE_IN) aS[t] = eattr[(size_t)eBase * EDGE_IN + t];
  __syncthreads();

  const int w = t >> 5;
  const int sub = t & 31;
  const int edge = eBase + w;
  const int c0 = sub * 4;
  const int src = ei[edge];
  const int dst = ei[N_EDGES + edge];

  float a0 = bS[c0], a1 = bS[c0 + 1], a2 = bS[c0 + 2], a3 = bS[c0 + 3];
#pragma unroll
  for (int k = 0; k < EDGE_IN; ++k) {
    const float av = aS[w * EDGE_IN + k];
    const float* wr = wS + k * D + c0;
    a0 += av * wr[0]; a1 += av * wr[1]; a2 += av * wr[2]; a3 += av * wr[3];
  }
  a0 = fmaxf(a0, 0.f); a1 = fmaxf(a1, 0.f);
  a2 = fmaxf(a2, 0.f); a3 = fmaxf(a3, 0.f);

  const float4 hv = *(const float4*)(h + (size_t)src * D + c0);
  float* ap = agg + (size_t)dst * D + c0;
  atomicAdd(ap + 0, hv.x + a0);
  atomicAdd(ap + 1, hv.y + a1);
  atomicAdd(ap + 2, hv.z + a2);
  atomicAdd(ap + 3, hv.w + a3);
}

// ---------------------------------------------------------------------------
// Fused 2-layer MLP via WMMA: out = relu(agg@W1 + b1)@W2 + b2  (pre-BN),
// plus per-channel sum/sumsq atomics for BatchNorm statistics.
// 8 waves per WG; each wave owns one 16-row tile. hid stays in LDS (f16).
// Fragment layouts follow CDNA5 ISA 7.12.2 (wave32).
// ---------------------------------------------------------------------------
__global__ void __launch_bounds__(256) mlp_wmma_kernel(
    const float* __restrict__ agg, const float* __restrict__ w1,
    const float* __restrict__ b1, const float* __restrict__ w2,
    const float* __restrict__ b2, float* __restrict__ outb,
    float* __restrict__ chanStats /* sum@[0,128), sumsq@[128,256) */) {
  extern __shared__ char smem[];
  half_t* w1t = (half_t*)smem;            // [n:256][k:128]
  half_t* w2t = w1t + D * D2;             // [n:128][k:256]
  float* b1s = (float*)(w2t + D * D2);
  float* b2s = b1s + D2;
  half_t* hid = (half_t*)(b2s + D);       // per-wave [16][256]

  const int t = threadIdx.x;
  for (int i = t; i < D * D2; i += 256) {  // W1: [128][256] row-major
    int k = i >> 8, n = i & 255;
    w1t[n * D + k] = (half_t)w1[i];
  }
  for (int i = t; i < D * D2; i += 256) {  // W2: [256][128] row-major
    int k = i >> 7, n = i & 127;
    w2t[n * D2 + k] = (half_t)w2[i];
  }
  if (t < D2) b1s[t] = b1[t];
  if (t < D) b2s[t] = b2[t];
  __syncthreads();

  const int wave = t >> 5, lane = t & 31;
  const int m = lane & 15;        // row (A/D) or column (B/D) index
  const int cg = lane >> 4;       // lane-half select
  const int c8 = cg * 8, c16 = cg * 16;
  half_t* myhid = hid + wave * (16 * D2);

  float csum[8], csq[8];
#pragma unroll
  for (int i = 0; i < 8; ++i) { csum[i] = 0.f; csq[i] = 0.f; }

  const int NT = N_NODES / 16;  // 1250, exact
  const int tile = blockIdx.x * 8 + wave;
  if (tile < NT) {
    const int rowBase = tile * 16;

    // ---- GEMM1: hid = relu(agg[16x128] @ W1[128x256] + b1) ----
    const float* aRow = agg + (size_t)(rowBase + m) * D;
    v16h afr[4];
#pragma unroll
    for (int kt = 0; kt < 4; ++kt) {
      const float* ap = aRow + kt * 32 + c8;
      v16h a;
#pragma unroll
      for (int hh = 0; hh < 8; ++hh) {
        a[hh] = (half_t)ap[hh];          // K = base + [0..7]
        a[hh + 8] = (half_t)ap[hh + 16]; // K = base + 16 + [0..7]
      }
      afr[kt] = a;
    }
#pragma unroll 1
    for (int nt = 0; nt < 16; ++nt) {
      const int n = nt * 16 + m;
      const half_t* wcol = w1t + n * D + c8;
      v8f acc = {};
#pragma unroll
      for (int kt = 0; kt < 4; ++kt) {
        v16h b;
        const half_t* bp = wcol + kt * 32;
#pragma unroll
        for (int hh = 0; hh < 8; ++hh) { b[hh] = bp[hh]; b[hh + 8] = bp[hh + 16]; }
        acc = __builtin_amdgcn_wmma_f32_16x16x32_f16(
            false, afr[kt], false, b, (short)0, acc, false, false);
      }
      const float bias = b1s[n];
#pragma unroll
      for (int j = 0; j < 8; ++j) {
        float v = fmaxf(acc[j] + bias, 0.f);  // D row = j + 8*cg
        myhid[(j + c8) * D2 + n] = (half_t)v;
      }
    }

    // ---- GEMM2: out = hid[16x256] @ W2[256x128] + b2 ----
    const half_t* hrow = myhid + m * D2;
    v16h afr2[8];
#pragma unroll
    for (int kt = 0; kt < 8; ++kt) {
      const half_t* ap = hrow + kt * 32 + c8;
      v16h a;
#pragma unroll
      for (int hh = 0; hh < 8; ++hh) {
        a[hh] = ap[hh];
        a[hh + 8] = ap[hh + 16];
      }
      afr2[kt] = a;
    }
#pragma unroll 1
    for (int nt = 0; nt < 8; ++nt) {
      const int n = nt * 16 + m;
      const half_t* wcol = w2t + n * D2 + c8;
      v8f acc = {};
#pragma unroll
      for (int kt = 0; kt < 8; ++kt) {
        v16h b;
        const half_t* bp = wcol + kt * 32;
#pragma unroll
        for (int hh = 0; hh < 8; ++hh) { b[hh] = bp[hh]; b[hh + 8] = bp[hh + 16]; }
        acc = __builtin_amdgcn_wmma_f32_16x16x32_f16(
            false, afr2[kt], false, b, (short)0, acc, false, false);
      }
      const float bias = b2s[n];
#pragma unroll
      for (int j = 0; j < 8; ++j) {
        float v = acc[j] + bias;
        outb[(size_t)(rowBase + j + c8) * D + n] = v;
        csum[nt] += v;
        csq[nt] += v * v;
      }
    }
  }
#pragma unroll
  for (int i = 0; i < 8; ++i) {
    atomicAdd(&chanStats[i * 16 + m], csum[i]);
    atomicAdd(&chanStats[128 + i * 16 + m], csq[i]);
  }
}

// ---------------------------------------------------------------------------
// Fold batch stats + gamma/beta into per-channel scale/shift.
// ---------------------------------------------------------------------------
__global__ void bn_stats_kernel(const float* __restrict__ stats,
                                const float* __restrict__ gamma,
                                const float* __restrict__ beta,
                                float* __restrict__ ss) {
  const int c = threadIdx.x;
  const float invN = 1.f / (float)N_NODES;
  const float mean = stats[c] * invN;
  const float var = stats[128 + c] * invN - mean * mean;
  const float rstd = rsqrtf(var + BN_EPS);
  const float sc = rstd * gamma[c];
  ss[c] = sc;
  ss[128 + c] = beta[c] - mean * sc;
}

// h_next = (relu?)(out*scale + shift);  jk += h_next
__global__ void __launch_bounds__(256) bn_apply_kernel(
    const float* __restrict__ outb, const float* __restrict__ ss,
    float* __restrict__ h, float* __restrict__ jk, int do_relu) {
  const int i = blockIdx.x * 256 + threadIdx.x;  // exactly N*D threads
  const int c = i & (D - 1);
  float v = outb[i] * ss[c] + ss[128 + c];
  if (do_relu) v = fmaxf(v, 0.f);
  h[i] = v;
  jk[i] += v;
}

extern "C" void kernel_launch(void* const* d_in, const int* in_sizes, int n_in,
                              void* d_out, int out_size, void* d_ws,
                              size_t ws_size, hipStream_t stream) {
  (void)in_sizes; (void)n_in; (void)out_size; (void)ws_size;
  const float* x     = (const float*)d_in[0];
  const int*   ei    = (const int*)d_in[1];
  const float* eattr = (const float*)d_in[2];
  const float* npw   = (const float*)d_in[3];
  const float* npb   = (const float*)d_in[4];
  const float* epw   = (const float*)d_in[5];
  const float* epb   = (const float*)d_in[6];
  const float* m1w   = (const float*)d_in[7];
  const float* m1b   = (const float*)d_in[8];
  const float* m2w   = (const float*)d_in[9];
  const float* m2b   = (const float*)d_in[10];
  const float* gam   = (const float*)d_in[11];
  const float* bet   = (const float*)d_in[12];
  float* jk = (float*)d_out;

  float* ws   = (float*)d_ws;
  float* h    = ws;                              // N*D
  float* agg  = h + (size_t)N_NODES * D;         // N*D
  float* outb = agg + (size_t)N_NODES * D;       // N*D
  float* stats = outb + (size_t)N_NODES * D;     // 256 (sum,sumsq) + 256 (scale,shift)

  node_proj_kernel<<<N_NODES, D, 0, stream>>>(x, npw, npb, h, jk);

  for (int l = 0; l < NLAYERS; ++l) {
    zero_kernel<<<(N_NODES * D + 255) / 256, 256, 0, stream>>>(agg, N_NODES * D);
    zero_kernel<<<1, 256, 0, stream>>>(stats, 256);
    edge_scatter_kernel<<<N_EDGES / 8, 256, 0, stream>>>(ei, eattr, epw, epb,
                                                         h, agg);
    mlp_wmma_kernel<<<(N_NODES / 16 + 7) / 8, 256, MLP_SMEM_BYTES, stream>>>(
        agg, m1w + (size_t)l * D * D2, m1b + (size_t)l * D2,
        m2w + (size_t)l * D * D2, m2b + (size_t)l * D, outb, stats);
    bn_stats_kernel<<<1, D, 0, stream>>>(stats, gam + (size_t)l * D,
                                         bet + (size_t)l * D, stats + 256);
    bn_apply_kernel<<<(N_NODES * D) / 256, 256, 0, stream>>>(
        outb, stats + 256, h, jk, (l != NLAYERS - 1) ? 1 : 0);
  }
}
